// LinearAttention_67637144977802
// MI455X (gfx1250) — matrix operand order repaired and checked by reading, not compiled, not run on hardware
//
#include <hip/hip_runtime.h>
#include <hip/hip_bf16.h>
#include <math.h>

typedef __bf16 bf16;
typedef __attribute__((ext_vector_type(16))) __bf16 v16bf;
typedef __attribute__((ext_vector_type(8)))  __bf16 v8bf;
typedef __attribute__((ext_vector_type(4)))  __bf16 v4bf;
typedef __attribute__((ext_vector_type(8)))  float  v8f;

typedef unsigned int v4u __attribute__((ext_vector_type(4)));
typedef int          v8i __attribute__((ext_vector_type(8)));
typedef int          v4i __attribute__((ext_vector_type(4)));

#define B_    16
#define DIM_  256
#define NH_   4
#define HD_   64
#define NPIX  4096   // 64*64

// ---------------------------------------------------------------------------
// WMMA (CDNA5 wave32, V_WMMA_F32_16X16X32_BF16)
// ---------------------------------------------------------------------------
__device__ __forceinline__ v8f wmma_bf16(v16bf a, v16bf b, v8f c) {
  return __builtin_amdgcn_wmma_f32_16x16x32_bf16(false, a, false, b, (short)0, c,
                                                 false, false);
}

// A fragment 16(M)x32(K) from row-major [M][K] tile (stride lda elems):
// two contiguous 16B runs per lane -> 2x ds_load_b128.
__device__ __forceinline__ v16bf frag_a(const bf16* p, int lda) {
  int lane = threadIdx.x & 31;
  const bf16* q = p + (lane & 15) * lda + ((lane >> 4) * 8);
  v8bf lo = *(const v8bf*)(q);
  v8bf hi = *(const v8bf*)(q + 16);
  return __builtin_shufflevector(lo, hi, 0, 1, 2, 3, 4, 5, 6, 7,
                                 8, 9, 10, 11, 12, 13, 14, 15);
}

// B fragment 32(K)x16(N) from K-major memory Bt[N][K] (stride ld elems):
// two contiguous 16B runs per lane (same k order as frag_a).
__device__ __forceinline__ v16bf frag_b(const bf16* p, int ld) {
  int lane = threadIdx.x & 31;
  const bf16* q = p + (lane & 15) * ld + ((lane >> 4) * 8);
  v8bf lo = *(const v8bf*)(q);
  v8bf hi = *(const v8bf*)(q + 16);
  return __builtin_shufflevector(lo, hi, 0, 1, 2, 3, 4, 5, 6, 7,
                                 8, 9, 10, 11, 12, 13, 14, 15);
}

// ---------------------------------------------------------------------------
// Tensor Data Mover: 2-D bf16 tile -> LDS. D# per CDNA5 ISA
// §8 (group0: count/lds/global/type=2; group1: data_size=2B, dims, stride).
// ---------------------------------------------------------------------------
#if defined(__has_builtin)
#if __has_builtin(__builtin_amdgcn_tensor_load_to_lds) && \
    __has_builtin(__builtin_amdgcn_s_wait_tensorcnt)
#define HAVE_TDM 1
#endif
#endif
#ifndef HAVE_TDM
#define HAVE_TDM 0
#endif

#if HAVE_TDM
__device__ __forceinline__ void tdm_load_tile_bf16(void* lds, const void* gaddr,
                                                   unsigned tw, unsigned th,
                                                   unsigned long long stride) {
  unsigned long long ga = (unsigned long long)(uintptr_t)gaddr;
  v4u g0;
  g0[0] = 1u;                                            // count=1, user desc
  g0[1] = (unsigned)(uintptr_t)lds;                      // lds_addr (bytes)
  g0[2] = (unsigned)ga;                                  // global_addr[31:0]
  g0[3] = (unsigned)((ga >> 32) & 0x01ffffffu) | (2u << 30);  // [56:32]|type=2
  v8i g1;
  g1[0] = (int)(1u << 16);                               // data_size=1 -> 2B
  g1[1] = (int)((tw & 0xffffu) << 16);                   // tensor_dim0 = tw
  g1[2] = (int)(((tw >> 16) & 0xffffu) | ((th & 0xffffu) << 16));  // dim1 = th
  g1[3] = (int)(((th >> 16) & 0xffffu) | ((tw & 0xffffu) << 16));  // tile_dim0
  g1[4] = (int)(th & 0xffffu);                           // tile_dim1
  g1[5] = (int)(stride & 0xffffffffu);                   // dim0_stride lo
  g1[6] = (int)((stride >> 32) & 0xffffu);               // dim0_stride hi
  g1[7] = 0;
  v4i z = {0, 0, 0, 0};
#if __clang_major__ >= 23
  v8i z8 = {0, 0, 0, 0, 0, 0, 0, 0};
  __builtin_amdgcn_tensor_load_to_lds(g0, g1, z, z, z8, 0);
#else
  __builtin_amdgcn_tensor_load_to_lds(g0, g1, z, z, 0);
#endif
}
#define TDM_WAIT() __builtin_amdgcn_s_wait_tensorcnt(0)
#endif

// ---------------------------------------------------------------------------
// P1: convert Wqkv (768x256) and Wproj (256x256) to bf16. grid 64, block 256.
// ---------------------------------------------------------------------------
__global__ __launch_bounds__(256) void k_cvt_w(const float* __restrict__ Wqkv,
                                               const float* __restrict__ Wproj,
                                               bf16* __restrict__ wq,
                                               bf16* __restrict__ wp) {
  const int idx = blockIdx.x * 256 + threadIdx.x;   // 16384 threads
#pragma unroll
  for (int i = 0; i < 3; ++i) {                     // 49152 float4 chunks
    int c = idx + i * 16384;
    float4 f = *(const float4*)&Wqkv[(size_t)c * 4];
    v4bf w; w[0] = (bf16)f.x; w[1] = (bf16)f.y; w[2] = (bf16)f.z; w[3] = (bf16)f.w;
    *(v4bf*)&wq[(size_t)c * 4] = w;
  }
  {
    float4 f = *(const float4*)&Wproj[(size_t)idx * 4];
    v4bf w; w[0] = (bf16)f.x; w[1] = (bf16)f.y; w[2] = (bf16)f.z; w[3] = (bf16)f.w;
    *(v4bf*)&wp[(size_t)idx * 4] = w;
  }
}

// ---------------------------------------------------------------------------
// P2: transpose+convert x[b][c][n] f32 -> xT[b][n][c] bf16 via LDS tile.
// grid (64 n-tiles, 4 c-tiles, 16 b), block 256.
// ---------------------------------------------------------------------------
__global__ __launch_bounds__(256) void k_xpose(const float* __restrict__ x,
                                               bf16* __restrict__ xT) {
  __shared__ float st[64][68];                      // +4 pad vs bank conflicts
  const int tid = threadIdx.x;
  const int b  = blockIdx.z;
  const int c0 = blockIdx.y * 64;
  const int n0 = blockIdx.x * 64;
#pragma unroll
  for (int i = 0; i < 4; ++i) {
    int idx = tid + i * 256;
    int c = idx >> 4, n4 = (idx & 15) * 4;
    float4 f = *(const float4*)&x[((size_t)b * DIM_ + c0 + c) * NPIX + n0 + n4];
    st[c][n4 + 0] = f.x; st[c][n4 + 1] = f.y;
    st[c][n4 + 2] = f.z; st[c][n4 + 3] = f.w;
  }
  if (tid < 64 && blockIdx.x + 1 < 64)              // global_prefetch_b8
    __builtin_prefetch(&x[((size_t)b * DIM_ + c0 + tid) * NPIX + n0 + 64], 0, 1);
  __syncthreads();
#pragma unroll
  for (int i = 0; i < 2; ++i) {
    int idx = tid + i * 256;
    int n = idx >> 3, c8 = (idx & 7) * 8;
    v8bf v;
#pragma unroll
    for (int j = 0; j < 8; ++j) v[j] = (bf16)st[c8 + j][n];
    *(v8bf*)&xT[((size_t)b * NPIX + n0 + n) * DIM_ + c0 + c8] = v;
  }
}

// ---------------------------------------------------------------------------
// K1: qkv = Wqkv * x_b, bf16 WMMA, TDM double-buffered tiles.
//  q: fused softmax over d, stored [b][h][n][d];  k: raw f32 [d][n];
//  v: /N bf16 [e][n].  grid (32 n-tiles, 12 o-tiles, 16 b), block 256.
// ---------------------------------------------------------------------------
__global__ __launch_bounds__(256) void k_qkv(const bf16* __restrict__ wq,
                                             const bf16* __restrict__ xT,
                                             bf16*  __restrict__ q_ws,
                                             float* __restrict__ k_raw,
                                             bf16*  __restrict__ v_ws) {
  __shared__ alignas(32) char smraw[32768];
  bf16* Ab[2] = {(bf16*)smraw, (bf16*)(smraw + 4096)};          // [64][32] x2
  bf16* Bb[2] = {(bf16*)(smraw + 8192), (bf16*)(smraw + 16384)}; // [128][32] x2
  float* soft = (float*)smraw;                                   // [64][128] f32

  const int tid = threadIdx.x;
  const int wv  = tid >> 5;
  const int b   = blockIdx.z;
  const int ot  = blockIdx.y;
  const int o0  = ot * 64;
  const int n0  = blockIdx.x * 128;

  const bf16* gA = wq + (size_t)o0 * DIM_;
  const bf16* gB = xT + ((size_t)b * NPIX + n0) * DIM_;

  v8f acc[4];
#pragma unroll
  for (int i = 0; i < 4; ++i) { v8f z = {}; acc[i] = z; }
  const int mt = wv >> 1;
  const int nq = (wv & 1) * 4;

  int cur = 0;
#if HAVE_TDM
  if (wv == 0) {
    tdm_load_tile_bf16(Ab[0], gA, 32, 64, DIM_);
    tdm_load_tile_bf16(Bb[0], gB, 32, 128, DIM_);
    TDM_WAIT();
  }
  __syncthreads();
#endif
  for (int kc = 0; kc < DIM_; kc += 32) {
    const bool more = (kc + 32 < DIM_);
#if HAVE_TDM
    if (more && wv == 0) {           // overlap next DMA with current WMMAs
      tdm_load_tile_bf16(Ab[cur ^ 1], gA + kc + 32, 32, 64, DIM_);
      tdm_load_tile_bf16(Bb[cur ^ 1], gB + kc + 32, 32, 128, DIM_);
    }
#else
#pragma unroll
    for (int idx = tid; idx < 256; idx += 256) {   // A 64x32
      int r = idx >> 2, c8 = (idx & 3) * 8;
      *(v8bf*)&Ab[cur][r * 32 + c8] = *(const v8bf*)&gA[(size_t)r * DIM_ + kc + c8];
    }
#pragma unroll
    for (int idx = tid; idx < 512; idx += 256) {   // B 128x32
      int n = idx >> 2, c8 = (idx & 3) * 8;
      *(v8bf*)&Bb[cur][n * 32 + c8] = *(const v8bf*)&gB[(size_t)n * DIM_ + kc + c8];
    }
    __syncthreads();
#endif
    v16bf a = frag_a(Ab[cur] + mt * 16 * 32, 32);
#pragma unroll
    for (int t = 0; t < 4; ++t) {
      v16bf bb = frag_b(Bb[cur] + (nq + t) * 16 * 32, 32);
      acc[t] = wmma_bf16(a, bb, acc[t]);
    }
#if HAVE_TDM
    if (more) {
      if (wv == 0) TDM_WAIT();
      __syncthreads();
    }
#else
    __syncthreads();
#endif
    cur ^= 1;
  }
  __syncthreads();                    // epilogue reuses the tile LDS region

  const int lane = tid & 31;
  const int nl   = lane & 15;
  const int mh   = (lane >> 4) * 8;
  const int sec  = ot >> 2;           // 0=q, 1=k, 2=v (uniform per block)
  const int h    = ot & 3;
  const size_t rowbase = (size_t)(b * NH_ + h) * HD_;

  if (sec == 1) {                     // k: raw f32 [d][n]
#pragma unroll
    for (int t = 0; t < 4; ++t) {
      int ncol = n0 + (nq + t) * 16 + nl;
#pragma unroll
      for (int r = 0; r < 8; ++r)
        k_raw[(rowbase + mt * 16 + mh + r) * NPIX + ncol] = acc[t][r];
    }
  } else if (sec == 2) {              // v: /N bf16 [e][n]
    const float s = 1.0f / (float)NPIX;
#pragma unroll
    for (int t = 0; t < 4; ++t) {
      int ncol = n0 + (nq + t) * 16 + nl;
#pragma unroll
      for (int r = 0; r < 8; ++r)
        v_ws[(rowbase + mt * 16 + mh + r) * NPIX + ncol] = (bf16)(acc[t][r] * s);
    }
  } else {                            // q: softmax over d, store [n][d]
#pragma unroll
    for (int t = 0; t < 4; ++t) {
      int ncol = (nq + t) * 16 + nl;
#pragma unroll
      for (int r = 0; r < 8; ++r)
        soft[(mt * 16 + mh + r) * 128 + ncol] = acc[t][r];
    }
    __syncthreads();
    if (tid < 128) {
      int col = tid;
      float mx = -1e30f;
      for (int d = 0; d < 64; ++d) mx = fmaxf(mx, soft[d * 128 + col]);
      float s = 0.f;
      for (int d = 0; d < 64; ++d) s += __expf(soft[d * 128 + col] - mx);
      float inv = 0.125f / s;         // HD^-0.5 = 1/8
      bf16* dst = q_ws + ((size_t)(b * NH_ + h) * NPIX + n0 + col) * HD_;
      for (int d8 = 0; d8 < 64; d8 += 8) {
        v8bf v;
#pragma unroll
        for (int j = 0; j < 8; ++j)
          v[j] = (bf16)(__expf(soft[(d8 + j) * 128 + col] - mx) * inv);
        *(v8bf*)&dst[d8] = v;
      }
    }
  }
}

// ---------------------------------------------------------------------------
// K2: k softmax over N=4096 per (b,h,d) row. grid 4096, block 256.
// ---------------------------------------------------------------------------
__global__ __launch_bounds__(256) void k_ksoftmax(const float* __restrict__ k_raw,
                                                  bf16* __restrict__ k_ws) {
  __shared__ float red[256];
  const int tid = threadIdx.x;
  const size_t row = blockIdx.x;
  const float* src = k_raw + row * NPIX;
  float vals[16];
  float mx = -1e30f;
#pragma unroll
  for (int i = 0; i < 16; ++i) { vals[i] = src[tid + 256 * i]; mx = fmaxf(mx, vals[i]); }
  red[tid] = mx; __syncthreads();
  for (int s = 128; s > 0; s >>= 1) {
    if (tid < s) red[tid] = fmaxf(red[tid], red[tid + s]);
    __syncthreads();
  }
  mx = red[0]; __syncthreads();
  float sum = 0.f;
#pragma unroll
  for (int i = 0; i < 16; ++i) { vals[i] = __expf(vals[i] - mx); sum += vals[i]; }
  red[tid] = sum; __syncthreads();
  for (int s = 128; s > 0; s >>= 1) {
    if (tid < s) red[tid] += red[tid + s];
    __syncthreads();
  }
  const float inv = 1.0f / red[0];
  bf16* dst = k_ws + row * NPIX;
#pragma unroll
  for (int i = 0; i < 16; ++i) dst[tid + 256 * i] = (bf16)(vals[i] * inv);
}

// ---------------------------------------------------------------------------
// K3: ctx[e][d] = sum_n k'[d][n] * v'[e][n], K=4096 per (b,h). grid 64.
// TDM double-buffered 64x64 tiles.
// ---------------------------------------------------------------------------
__global__ __launch_bounds__(256) void k_ctx(const bf16* __restrict__ k_ws,
                                             const bf16* __restrict__ v_ws,
                                             bf16* __restrict__ ctx_ws) {
  __shared__ alignas(32) char smraw[32768];
  bf16* kb[2] = {(bf16*)smraw, (bf16*)(smraw + 8192)};
  bf16* vb[2] = {(bf16*)(smraw + 16384), (bf16*)(smraw + 24576)};
  const int tid = threadIdx.x;
  const int wv  = tid >> 5;
  const int bh  = blockIdx.x;
  const bf16* gk = k_ws + (size_t)bh * HD_ * NPIX;
  const bf16* gv = v_ws + (size_t)bh * HD_ * NPIX;

  v8f acc[2];
  { v8f z = {}; acc[0] = z; acc[1] = z; }
  const int t0 = wv * 2;
  const int tm = t0 >> 2;

  int cur = 0;
#if HAVE_TDM
  if (wv == 0) {
    tdm_load_tile_bf16(kb[0], gk, 64, 64, NPIX);
    tdm_load_tile_bf16(vb[0], gv, 64, 64, NPIX);
    TDM_WAIT();
  }
  __syncthreads();
#endif
  for (int nc = 0; nc < NPIX; nc += 64) {
    const bool more = (nc + 64 < NPIX);
#if HAVE_TDM
    if (more && wv == 0) {
      tdm_load_tile_bf16(kb[cur ^ 1], gk + nc + 64, 64, 64, NPIX);
      tdm_load_tile_bf16(vb[cur ^ 1], gv + nc + 64, 64, 64, NPIX);
    }
#else
#pragma unroll
    for (int i = tid; i < 512; i += 256) {
      int r = i >> 3, c = (i & 7) * 8;
      *(v8bf*)&kb[cur][r * 64 + c] = *(const v8bf*)&gk[(size_t)r * NPIX + nc + c];
      *(v8bf*)&vb[cur][r * 64 + c] = *(const v8bf*)&gv[(size_t)r * NPIX + nc + c];
    }
    __syncthreads();
#endif
#pragma unroll
    for (int ks = 0; ks < 2; ++ks) {
      v16bf a = frag_a(kb[cur] + tm * 16 * 64 + ks * 32, 64);
#pragma unroll
      for (int i = 0; i < 2; ++i) {
        int tn = (t0 + i) & 3;
        v16bf bb = frag_b(vb[cur] + tn * 16 * 64 + ks * 32, 64);
        acc[i] = wmma_bf16(a, bb, acc[i]);
      }
    }
#if HAVE_TDM
    if (more) {
      if (wv == 0) TDM_WAIT();
      __syncthreads();
    }
#else
    __syncthreads();
#endif
    cur ^= 1;
  }

  const int lane = tid & 31;
  const int nl   = lane & 15;
  const int mh   = (lane >> 4) * 8;
  bf16* dst = ctx_ws + (size_t)bh * HD_ * HD_;      // stored [e][d]
#pragma unroll
  for (int i = 0; i < 2; ++i) {
    int tn = (t0 + i) & 3;
    v8bf v;
#pragma unroll
    for (int r = 0; r < 8; ++r) v[r] = (bf16)acc[i][r];
    *(v8bf*)&dst[(tn * 16 + nl) * 64 + tm * 16 + mh] = v;
  }
}

// ---------------------------------------------------------------------------
// K4: out[e][n] = sum_d ctx[d][e] * q'[d][n]; out -> [b][n][c] bf16.
// grid (32 n-tiles, 64 bh), block 256.
// ---------------------------------------------------------------------------
__global__ __launch_bounds__(256) void k_out(const bf16* __restrict__ ctx_ws,
                                             const bf16* __restrict__ q_ws,
                                             bf16* __restrict__ out_t) {
  __shared__ alignas(32) bf16 cs[64 * 64];    // ctx [e][d]
  __shared__ alignas(32) bf16 qt[128 * 64];   // q'  [n][d]
  const int tid = threadIdx.x;
  const int wv  = tid >> 5;
  const int bh  = blockIdx.y;
  const int b   = bh >> 2, h = bh & 3;
  const int n0  = blockIdx.x * 128;
  const bf16* gq = q_ws + (size_t)bh * NPIX * HD_;

#if HAVE_TDM
  if (wv == 0) {
    tdm_load_tile_bf16(cs, ctx_ws + (size_t)bh * 4096, 64, 64, 64);
    tdm_load_tile_bf16(qt, gq + (size_t)n0 * HD_, 64, 128, 64);
    TDM_WAIT();
  }
  __syncthreads();
#else
#pragma unroll
  for (int i = tid; i < 512; i += 256) {
    int r = i >> 3, c = (i & 7) * 8;
    *(v8bf*)&cs[r * 64 + c] = *(const v8bf*)&ctx_ws[(size_t)bh * 4096 + r * 64 + c];
  }
#pragma unroll
  for (int i = tid; i < 1024; i += 256) {
    int n = i >> 3, c = (i & 7) * 8;
    *(v8bf*)&qt[n * 64 + c] = *(const v8bf*)&gq[(size_t)(n0 + n) * HD_ + c];
  }
  __syncthreads();
#endif

  const int mt = wv >> 1;
  const int nq = (wv & 1) * 4;
  v8f acc[4];
#pragma unroll
  for (int i = 0; i < 4; ++i) { v8f z = {}; acc[i] = z; }

#pragma unroll
  for (int ks = 0; ks < 2; ++ks) {
    v16bf a = frag_a(cs + mt * 16 * 64 + ks * 32, 64);
#pragma unroll
    for (int t = 0; t < 4; ++t) {
      v16bf bb = frag_b(qt + (nq + t) * 16 * 64 + ks * 32, 64);
      acc[t] = wmma_bf16(a, bb, acc[t]);
    }
  }

  const int lane = tid & 31;
  const int nl   = lane & 15;
  const int mh   = (lane >> 4) * 8;
  bf16* outp = out_t + (size_t)b * NPIX * DIM_;
#pragma unroll
  for (int t = 0; t < 4; ++t) {
    int n = n0 + (nq + t) * 16 + nl;
    v8bf v;
#pragma unroll
    for (int r = 0; r < 8; ++r) v[r] = (bf16)acc[t][r];
    *(v8bf*)&outp[(size_t)n * DIM_ + h * 64 + mt * 16 + mh] = v;
  }
}

// ---------------------------------------------------------------------------
// K5: y = Wproj*out + bproj, LayerNorm over 256 channels, * gamma.
// TDM double-buffered tiles. grid (64 n-tiles, 16 b), block 256.
// ---------------------------------------------------------------------------
__global__ __launch_bounds__(256) void k_proj_ln(const bf16* __restrict__ out_t,
                                                 const bf16* __restrict__ wp,
                                                 const float* __restrict__ bproj,
                                                 const float* __restrict__ gamma,
                                                 float* __restrict__ y) {
  __shared__ alignas(32) char smraw[66048];
  bf16* Wb[2] = {(bf16*)smraw, (bf16*)(smraw + 16384)};           // [256][32] x2
  bf16* Ob[2] = {(bf16*)(smraw + 32768), (bf16*)(smraw + 36864)}; // [64][32] x2
  float* Y    = (float*)smraw;             // [256][64] f32 (reused post-loop)
  float* stat = (float*)(smraw + 65536);   // mean[64], rstd[64]

  const int tid = threadIdx.x;
  const int wv  = tid >> 5;
  const int b   = blockIdx.y;
  const int n0  = blockIdx.x * 64;
  const bf16* gO = out_t + ((size_t)b * NPIX + n0) * DIM_;

  v8f acc[8];
#pragma unroll
  for (int i = 0; i < 8; ++i) { v8f z = {}; acc[i] = z; }
  const int mt0 = wv * 2;

  int cur = 0;
#if HAVE_TDM
  if (wv == 0) {
    tdm_load_tile_bf16(Wb[0], wp, 32, 256, DIM_);
    tdm_load_tile_bf16(Ob[0], gO, 32, 64, DIM_);
    TDM_WAIT();
  }
  __syncthreads();
#endif
  for (int kc = 0; kc < DIM_; kc += 32) {
    const bool more = (kc + 32 < DIM_);
#if HAVE_TDM
    if (more && wv == 0) {
      tdm_load_tile_bf16(Wb[cur ^ 1], wp + kc + 32, 32, 256, DIM_);
      tdm_load_tile_bf16(Ob[cur ^ 1], gO + kc + 32, 32, 64, DIM_);
    }
#else
#pragma unroll
    for (int i = tid; i < 1024; i += 256) {        // Wproj 256x32
      int r = i >> 2, c8 = (i & 3) * 8;
      *(v8bf*)&Wb[cur][r * 32 + c8] = *(const v8bf*)&wp[(size_t)r * DIM_ + kc + c8];
    }
#pragma unroll
    for (int i = tid; i < 256; i += 256) {         // out tile 64x32
      int n = i >> 2, c8 = (i & 3) * 8;
      *(v8bf*)&Ob[cur][n * 32 + c8] = *(const v8bf*)&gO[(size_t)n * DIM_ + kc + c8];
    }
    __syncthreads();
#endif
#pragma unroll
    for (int im = 0; im < 2; ++im) {
      v16bf a = frag_a(Wb[cur] + (mt0 + im) * 16 * 32, 32);
#pragma unroll
      for (int t = 0; t < 4; ++t) {
        v16bf bb = frag_b(Ob[cur] + t * 16 * 32, 32);
        acc[im * 4 + t] = wmma_bf16(a, bb, acc[im * 4 + t]);
      }
    }
#if HAVE_TDM
    if (more) {
      if (wv == 0) TDM_WAIT();
      __syncthreads();
    }
#else
    __syncthreads();
#endif
    cur ^= 1;
  }
  __syncthreads();                      // Y aliases the tile buffers

  const int lane = tid & 31;
  const int nl   = lane & 15;
  const int mh   = (lane >> 4) * 8;
#pragma unroll
  for (int im = 0; im < 2; ++im)
#pragma unroll
    for (int t = 0; t < 4; ++t)
#pragma unroll
      for (int r = 0; r < 8; ++r) {
        int o = (mt0 + im) * 16 + mh + r;
        Y[o * 64 + t * 16 + nl] = acc[im * 4 + t][r] + bproj[o];
      }
  __syncthreads();

  float* meanA = stat;
  float* rstdA = stat + 64;
  if (tid < 64) {
    float s = 0.f, s2 = 0.f;
    for (int o = 0; o < 256; ++o) { float v = Y[o * 64 + tid]; s += v; s2 += v * v; }
    float mean = s * (1.0f / 256.0f);
    float var  = s2 * (1.0f / 256.0f) - mean * mean;
    meanA[tid] = mean;
    rstdA[tid] = rsqrtf(var + 1e-5f);
  }
  __syncthreads();

  for (int i = tid; i < 256 * 64; i += 256) {
    int o = i >> 6, n = i & 63;
    y[((size_t)b * DIM_ + o) * NPIX + n0 + n] =
        (Y[i] - meanA[n]) * rstdA[n] * gamma[o];
  }
}

// ---------------------------------------------------------------------------
extern "C" void kernel_launch(void* const* d_in, const int* in_sizes, int n_in,
                              void* d_out, int out_size, void* d_ws, size_t ws_size,
                              hipStream_t stream) {
  (void)in_sizes; (void)n_in; (void)out_size; (void)ws_size;
  const float* x     = (const float*)d_in[0];
  const float* Wqkv  = (const float*)d_in[1];
  const float* Wproj = (const float*)d_in[2];
  const float* bproj = (const float*)d_in[3];
  const float* gamma = (const float*)d_in[4];
  float* y = (float*)d_out;

  char* ws = (char*)d_ws;
  size_t off = 0;
  const size_t NELEM = (size_t)B_ * DIM_ * NPIX;
  bf16*  q_ws   = (bf16*)(ws + off);  off += NELEM * 2;   // q' [b][h][n][d]
  float* k_raw  = (float*)(ws + off); off += NELEM * 4;   // raw k; later out_t
  bf16*  k_ws   = (bf16*)(ws + off);  off += NELEM * 2;   // k' [d][n]
  bf16*  v_ws   = (bf16*)(ws + off);  off += NELEM * 2;   // v/N [e][n]
  bf16*  xT     = (bf16*)(ws + off);  off += NELEM * 2;   // x^T [b][n][c]
  bf16*  ctx_ws = (bf16*)(ws + off);  off += (size_t)B_ * NH_ * HD_ * HD_ * 2;
  bf16*  wq_bf  = (bf16*)(ws + off);  off += (size_t)3 * DIM_ * DIM_ * 2;
  bf16*  wp_bf  = (bf16*)(ws + off);  off += (size_t)DIM_ * DIM_ * 2;
  bf16*  out_t  = (bf16*)k_raw;       // reuse dead k_raw region: [b][n][c]

  k_cvt_w<<<dim3(64), 256, 0, stream>>>(Wqkv, Wproj, wq_bf, wp_bf);
  k_xpose<<<dim3(64, 4, 16), 256, 0, stream>>>(x, xT);
  k_qkv<<<dim3(32, 12, 16), 256, 0, stream>>>(wq_bf, xT, q_ws, k_raw, v_ws);
  k_ksoftmax<<<dim3(4096), 256, 0, stream>>>(k_raw, k_ws);
  k_ctx<<<dim3(64), 256, 0, stream>>>(k_ws, v_ws, ctx_ws);
  k_out<<<dim3(32, 64), 256, 0, stream>>>(ctx_ws, q_ws, out_t);
  k_proj_ln<<<dim3(64, 16), 256, 0, stream>>>(out_t, wp_bf, bproj, gamma, y);
}
